// PCAReducer_9878424781335
// MI455X (gfx1250) — matrix operand order repaired and checked by reading, not compiled, not run on hardware
//
#include <hip/hip_runtime.h>

typedef __attribute__((ext_vector_type(16))) __bf16 v16bf;
typedef __attribute__((ext_vector_type(8)))  float  v8f;

#define BATCH 16384
#define FEAT  2048

// ---------- bf16 helpers (RNE split: v = hi + lo) ----------
static __device__ inline unsigned short f2bf_bits(float f) {
    union { float f; unsigned u; } a; a.f = f;
    unsigned r = a.u + 0x7FFFu + ((a.u >> 16) & 1u);
    return (unsigned short)(r >> 16);
}
static __device__ inline float bfbits2f(unsigned short h) {
    union { unsigned u; float f; } a; a.u = ((unsigned)h) << 16;
    return a.f;
}

// ---------- row means (float4 vectorized) ----------
__global__ void __launch_bounds__(256) mean_kernel(const float* __restrict__ x,
                                                   float* __restrict__ mean) {
    int row  = blockIdx.x * 8 + (threadIdx.x >> 5);
    int lane = threadIdx.x & 31;
    const float4* p = (const float4*)(x + (size_t)row * FEAT);
    float s = 0.0f;
    for (int f4 = lane; f4 < FEAT / 4; f4 += 32) {
        float4 v = p[f4];
        s += (v.x + v.y) + (v.z + v.w);
    }
    for (int off = 16; off; off >>= 1) s += __shfl_down(s, off, 32);
    if (lane == 0) mean[row] = s * (1.0f / (float)FEAT);
}

// ---------- fragment loaders from feature-major LDS tiles [feat][40] ----------
// A 16x32 bf16: lanes 0-15 (M=lane): K 0..7 then 16..23; lanes 16-31: K 8..15 then 24..31
static __device__ inline v16bf load_frag_A(const __bf16 (*T)[40], int fbase, int lane) {
    int fl = fbase + (lane & 15);
    int kb = (lane < 16) ? 0 : 8;
    v16bf r;
    ((uint4*)&r)[0] = *(const uint4*)&T[fl][kb];
    ((uint4*)&r)[1] = *(const uint4*)&T[fl][kb + 16];
    return r;
}
// B 32x16 bf16: lanes 0-15 (N=lane): K 0..7 then 16..23; lanes 16-31: K 8..15 then 24..31
static __device__ inline v16bf load_frag_B(const __bf16 (*T)[40], int fbase, int lane) {
    int fl = fbase + (lane & 15);
    int kb = (lane < 16) ? 0 : 8;
    v16bf r;
    ((uint4*)&r)[0] = *(const uint4*)&T[fl][kb];
    ((uint4*)&r)[1] = *(const uint4*)&T[fl][kb + 16];
    return r;
}

// Stage one 128-feature x 32-k tile: thread handles 4 feats x 4 k,
// packs 4 k-consecutive bf16 into one ds_store_b64 per plane.
static __device__ inline void stage_tile(const float* __restrict__ x,
                                         const float* __restrict__ mm,
                                         int k0, int fbase_global,
                                         int fg, int kg,
                                         __bf16 (*Thi)[40], __bf16 (*Tlo)[40]) {
    float4 xv[4];
    for (int kk = 0; kk < 4; ++kk)
        xv[kk] = *(const float4*)&x[(size_t)(k0 + kg + kk) * FEAT + fbase_global + fg];
    for (int ff = 0; ff < 4; ++ff) {
        unsigned short hs[4], ls[4];
        for (int kk = 0; kk < 4; ++kk) {
            float v = ((const float*)&xv[kk])[ff] - mm[kk];
            unsigned short hh = f2bf_bits(v);
            hs[kk] = hh;
            ls[kk] = f2bf_bits(v - bfbits2f(hh));
        }
        uint2 ph, pl;
        ph.x = (unsigned)hs[0] | ((unsigned)hs[1] << 16);
        ph.y = (unsigned)hs[2] | ((unsigned)hs[3] << 16);
        pl.x = (unsigned)ls[0] | ((unsigned)ls[1] << 16);
        pl.y = (unsigned)ls[2] | ((unsigned)ls[3] << 16);
        *(uint2*)&Thi[fg + ff][kg] = ph;
        *(uint2*)&Tlo[fg + ff][kg] = pl;
    }
}

// ---------- G = xc^T xc via split-bf16 WMMA ----------
// grid (16,16): block computes G[i0:i0+128, j0:j0+128]
// 8 waves as 4(m) x 2(n): wave tile 32x64 = 2x4 tiles of 16x16
__global__ void __launch_bounds__(256) gram_kernel(const float* __restrict__ x,
                                                   const float* __restrict__ mean,
                                                   float* __restrict__ G) {
    __shared__ __align__(16) __bf16 Ahi[128][40];
    __shared__ __align__(16) __bf16 Alo[128][40];
    __shared__ __align__(16) __bf16 Bhi[128][40];
    __shared__ __align__(16) __bf16 Blo[128][40];

    const int i0   = blockIdx.y * 128;
    const int j0   = blockIdx.x * 128;
    const int tid  = threadIdx.x;
    const int wave = tid >> 5;
    const int lane = tid & 31;
    const int mW   = (wave >> 1) * 32;   // 0,32,64,96
    const int nW   = (wave & 1) * 64;    // 0,64

    const int fg = (tid & 31) * 4;       // 0..124 feature within tile
    const int kg = (tid >> 5) * 4;       // 0..28  k within 32-slab

    v8f c[2][4] = {};

    for (int k0 = 0; k0 < BATCH; k0 += 32) {
        __syncthreads();
        float mm[4];
        for (int kk = 0; kk < 4; ++kk) mm[kk] = mean[k0 + kg + kk];
        stage_tile(x, mm, k0, i0, fg, kg, Ahi, Alo);
        stage_tile(x, mm, k0, j0, fg, kg, Bhi, Blo);
        if (k0 + 32 < BATCH) {  // prefetch next K-slab (global_prefetch_b8)
            __builtin_prefetch(&x[(size_t)(k0 + 32 + kg) * FEAT + i0 + fg], 0, 1);
            __builtin_prefetch(&x[(size_t)(k0 + 32 + kg) * FEAT + j0 + fg], 0, 1);
        }
        __syncthreads();

        // hoist all B fragments (reused across both mt iterations)
        v16bf bh[4], bl[4];
        for (int nt = 0; nt < 4; ++nt) {
            bh[nt] = load_frag_B(Bhi, nW + nt * 16, lane);
            bl[nt] = load_frag_B(Blo, nW + nt * 16, lane);
        }
        for (int mt = 0; mt < 2; ++mt) {
            v16bf ah = load_frag_A(Ahi, mW + mt * 16, lane);
            v16bf al = load_frag_A(Alo, mW + mt * 16, lane);
            for (int nt = 0; nt < 4; ++nt) {
                // G += hi*hi + hi*lo + lo*hi  (fp32 accumulate)
                c[mt][nt] = __builtin_amdgcn_wmma_f32_16x16x32_bf16(
                    false, ah, false, bh[nt], (short)0, c[mt][nt], false, false);
                c[mt][nt] = __builtin_amdgcn_wmma_f32_16x16x32_bf16(
                    false, ah, false, bl[nt], (short)0, c[mt][nt], false, false);
                c[mt][nt] = __builtin_amdgcn_wmma_f32_16x16x32_bf16(
                    false, al, false, bh[nt], (short)0, c[mt][nt], false, false);
            }
        }
    }

    // C/D layout: lanes 0-15: VGPR v -> M=v; lanes 16-31: VGPR v -> M=v+8; N=lane&15
    for (int mt = 0; mt < 2; ++mt)
        for (int nt = 0; nt < 4; ++nt) {
            int gm0 = i0 + mW + mt * 16 + ((lane < 16) ? 0 : 8);
            int gn  = j0 + nW + nt * 16 + (lane & 15);
            for (int v = 0; v < 8; ++v)
                G[(size_t)(gm0 + v) * FEAT + gn] = c[mt][nt][v];
        }
}

// ---------- deterministic init of V (2 columns, column-major) ----------
__global__ void init_v_kernel(float* __restrict__ V) {
    int f = blockIdx.x * 256 + threadIdx.x;   // 0..2047
    unsigned u1 = (unsigned)f * 2654435761u + 12345u;
    u1 ^= u1 >> 13; u1 *= 2246822519u; u1 ^= u1 >> 16;
    unsigned u2 = ((unsigned)f + 7919u) * 2654435761u + 777u;
    u2 ^= u2 >> 13; u2 *= 2246822519u; u2 ^= u2 >> 16;
    V[f]        = (float)(u1 & 0xFFFFu) * (1.0f / 65536.0f) - 0.5f;
    V[f + FEAT] = (float)(u2 & 0xFFFFu) * (1.0f / 65536.0f) - 0.5f;
}

// ---------- W = G @ V (wave per row; G stays hot in L2; float4) ----------
__global__ void __launch_bounds__(256) gemv2_kernel(const float* __restrict__ G,
                                                    const float* __restrict__ V,
                                                    float* __restrict__ W) {
    int row  = blockIdx.x * 8 + (threadIdx.x >> 5);
    int lane = threadIdx.x & 31;
    const float4* g = (const float4*)(G + (size_t)row * FEAT);
    float a0 = 0.0f, a1 = 0.0f;
    for (int f4 = lane; f4 < FEAT / 4; f4 += 32) {
        float4 gv = g[f4];
        float4 v0 = *(const float4*)&V[f4 * 4];
        float4 v1 = *(const float4*)&V[f4 * 4 + FEAT];
        a0 += gv.x * v0.x + gv.y * v0.y + gv.z * v0.z + gv.w * v0.w;
        a1 += gv.x * v1.x + gv.y * v1.y + gv.z * v1.z + gv.w * v1.w;
    }
    for (int off = 16; off; off >>= 1) {
        a0 += __shfl_down(a0, off, 32);
        a1 += __shfl_down(a1, off, 32);
    }
    if (lane == 0) { W[row] = a0; W[row + FEAT] = a1; }
}

// ---------- Gram-Schmidt orthonormalize (single block) ----------
__global__ void __launch_bounds__(256) ortho_kernel(const float* __restrict__ W,
                                                    float* __restrict__ V) {
    __shared__ float red[256];
    int t = threadIdx.x;
    float w1[8], w2[8];
    for (int i = 0; i < 8; ++i) {
        int f = t + i * 256;
        w1[i] = W[f];
        w2[i] = W[f + FEAT];
    }
    float s = 0.0f;
    for (int i = 0; i < 8; ++i) s += w1[i] * w1[i];
    red[t] = s; __syncthreads();
    for (int o = 128; o; o >>= 1) { if (t < o) red[t] += red[t + o]; __syncthreads(); }
    float rn1 = rsqrtf(red[0] + 1e-30f);
    __syncthreads();
    float d = 0.0f;
    for (int i = 0; i < 8; ++i) { w1[i] *= rn1; d += w1[i] * w2[i]; }
    red[t] = d; __syncthreads();
    for (int o = 128; o; o >>= 1) { if (t < o) red[t] += red[t + o]; __syncthreads(); }
    float dot = red[0];
    __syncthreads();
    float s2 = 0.0f;
    for (int i = 0; i < 8; ++i) { w2[i] -= dot * w1[i]; s2 += w2[i] * w2[i]; }
    red[t] = s2; __syncthreads();
    for (int o = 128; o; o >>= 1) { if (t < o) red[t] += red[t + o]; __syncthreads(); }
    float rn2 = rsqrtf(red[0] + 1e-30f);
    for (int i = 0; i < 8; ++i) {
        int f = t + i * 256;
        V[f]        = w1[i];
        V[f + FEAT] = w2[i] * rn2;
    }
}

// ---------- reduced = xc @ V (bandwidth bound; wave per row; float4) ----------
__global__ void __launch_bounds__(256) proj_kernel(const float* __restrict__ x,
                                                   const float* __restrict__ mean,
                                                   const float* __restrict__ V,
                                                   float* __restrict__ out) {
    int row  = blockIdx.x * 8 + (threadIdx.x >> 5);
    int lane = threadIdx.x & 31;
    const float4* p = (const float4*)(x + (size_t)row * FEAT);
    float m  = mean[row];
    float a0 = 0.0f, a1 = 0.0f;
    for (int f4 = lane; f4 < FEAT / 4; f4 += 32) {
        float4 xv = p[f4];
        float4 v0 = *(const float4*)&V[f4 * 4];
        float4 v1 = *(const float4*)&V[f4 * 4 + FEAT];
        float cx = xv.x - m, cy = xv.y - m, cz = xv.z - m, cw = xv.w - m;
        a0 += cx * v0.x + cy * v0.y + cz * v0.z + cw * v0.w;
        a1 += cx * v1.x + cy * v1.y + cz * v1.z + cw * v1.w;
    }
    for (int off = 16; off; off >>= 1) {
        a0 += __shfl_down(a0, off, 32);
        a1 += __shfl_down(a1, off, 32);
    }
    if (lane == 0) {
        out[(size_t)row * 2 + 0] = a0;
        out[(size_t)row * 2 + 1] = a1;
    }
}

extern "C" void kernel_launch(void* const* d_in, const int* in_sizes, int n_in,
                              void* d_out, int out_size, void* d_ws, size_t ws_size,
                              hipStream_t stream) {
    (void)in_sizes; (void)n_in; (void)out_size; (void)ws_size;
    const float* x   = (const float*)d_in[0];
    float*       out = (float*)d_out;

    // workspace layout (floats): mean[16384] | G[2048*2048] | V[2*2048] | W[2*2048]
    float* ws   = (float*)d_ws;
    float* mean = ws;
    float* G    = ws + BATCH;
    float* V    = G + (size_t)FEAT * FEAT;
    float* W    = V + 2 * FEAT;

    mean_kernel<<<BATCH / 8, 256, 0, stream>>>(x, mean);

    dim3 gg(FEAT / 128, FEAT / 128);
    gram_kernel<<<gg, 256, 0, stream>>>(x, mean, G);

    init_v_kernel<<<FEAT / 256, 256, 0, stream>>>(V);
    for (int it = 0; it < 48; ++it) {
        gemv2_kernel<<<FEAT / 8, 256, 0, stream>>>(G, V, W);
        ortho_kernel<<<1, 256, 0, stream>>>(W, V);
    }

    proj_kernel<<<BATCH / 8, 256, 0, stream>>>(x, mean, V, out);
}
